// ElmanRNN_sparse_10728828305918
// MI455X (gfx1250) — matrix operand order repaired and checked by reading, not compiled, not run on hardware
//
#include <hip/hip_runtime.h>
#include <hip/hip_bf16.h>

// Problem dims (static)
#define BB    16
#define SS    512
#define DIN   1024
#define HH    1024
#define DOUT  512
#define ROWS  (BB*SS)          // 8192
#define NWG_R 64               // recurrence workgroups (one per 16-col N-tile of H)

typedef __attribute__((ext_vector_type(16))) __bf16 v16bf;
typedef __attribute__((ext_vector_type(8)))  float  v8f;

// ---------------------------------------------------------------------------
// Packed B layout for V_WMMA_F32_16X16X32_BF16:
//   B is 32x16 (KxN) per tile. Lane l holds B[Kbase+i][n], i=0..15,
//   n = l&15, Kbase = (l>>4)*16.  We pack so each lane's 16 bf16 are
//   contiguous (32B):  off(nt,kt,n,khalf) = (nt*KT+kt)*512 + n*32 + khalf*16
//   and value = W[nt*16+n][kt*32 + khalf*16 + i]   (B = W^T)
// ---------------------------------------------------------------------------
__global__ void k_pack_w(const float* __restrict__ W, __bf16* __restrict__ P,
                         int R, int K) {
  int idx = blockIdx.x * blockDim.x + threadIdx.x;     // one per 16-elem chunk
  int chunks = (R * K) >> 4;
  if (idx >= chunks) return;
  int KT    = K >> 5;
  int khalf = idx & 1;
  int n     = (idx >> 1) & 15;
  int tkt   = idx >> 5;             // nt*KT + kt
  int kt    = tkt % KT;
  int nt    = tkt / KT;
  int j     = nt * 16 + n;
  int kbase = kt * 32 + khalf * 16;
  const float* src = W + (size_t)j * K + kbase;
  __bf16* dst = P + (size_t)idx * 16;
#pragma unroll
  for (int i = 0; i < 16; ++i) dst[i] = (__bf16)src[i];
}

// A-fragment loader (fp32 source -> bf16 frag), row-major source.
// Lane l: m = l&15, g = l>>4; elems i : K = kofs + g*16 + i
__device__ __forceinline__ v16bf load_a_f32(const float* __restrict__ base,
                                            size_t ldRow, int kofs, int lane) {
  int m = lane & 15, g = lane >> 4;
  const float* p0 = base + (size_t)m * ldRow + kofs + g * 16;
  v16bf a;
#pragma unroll
  for (int i = 0; i < 8; ++i) a[i] = (__bf16)p0[i];
#pragma unroll
  for (int i = 0; i < 8; ++i) a[8 + i] = (__bf16)p0[8 + i];
  return a;
}

__device__ __forceinline__ v16bf load_b_pk(const __bf16* __restrict__ tile, int lane) {
  return *(const v16bf*)(tile + (lane & 15) * 32 + (lane >> 4) * 16);
}

// ---------------------------------------------------------------------------
// Phase 1: Apre[r][h] = x[r]@W_ih^T + b_ih + b_hh      (r = b*S+s, 8192x1024)
// grid 512 blocks (one M-tile) x 256 threads (8 waves); wave w -> n-tiles w*8..w*8+7
// ---------------------------------------------------------------------------
__global__ void k_gemm_in(const float* __restrict__ x, const __bf16* __restrict__ pW,
                          const float* __restrict__ b_ih, const float* __restrict__ b_hh,
                          float* __restrict__ Apre) {
  const int mt = blockIdx.x;
  const int lane = threadIdx.x & 31;
  const int wave = threadIdx.x >> 5;
  const int g = lane >> 4;
  const float* arow = x + (size_t)(mt * 16 + (lane & 15)) * DIN;
  v8f acc[8] = {};
  for (int kt = 0; kt < DIN / 32; ++kt) {
    v16bf a = load_a_f32(x + (size_t)(mt * 16) * DIN, DIN, kt * 32, lane);
    if (kt < DIN / 32 - 1)
      __builtin_prefetch(arow + (kt + 1) * 32, 0, 1);
#pragma unroll
    for (int j = 0; j < 8; ++j) {
      int nt = wave * 8 + j;
      v16bf b = load_b_pk(pW + (size_t)(nt * (DIN / 32) + kt) * 512, lane);
      acc[j] = __builtin_amdgcn_wmma_f32_16x16x32_bf16(false, a, false, b,
                                                       (short)0, acc[j], false, false);
    }
  }
#pragma unroll
  for (int j = 0; j < 8; ++j) {
    int nt = wave * 8 + j;
    int n = nt * 16 + (lane & 15);
    float bias = b_ih[n] + b_hh[n];
#pragma unroll
    for (int r = 0; r < 8; ++r) {
      int mm = mt * 16 + r + 8 * g;
      Apre[(size_t)mm * HH + n] = acc[j][r] + bias;
    }
  }
}

// ---------------------------------------------------------------------------
// Phase 2: persistent recurrence.  Z rows are (b*S + t) so Z doubles as the
// h-history; h_{t-1} is read from Z rows (b*S + t - 1).  W_hh column block
// for this workgroup's n-tile lives in LDS for all 512 steps.
// grid 64 blocks x 128 threads (4 waves); wave w does k-tiles w*8..w*8+7.
// ---------------------------------------------------------------------------
__global__ void k_recur(const float* __restrict__ Apre, const __bf16* __restrict__ pWhh,
                        float* __restrict__ Z, float* __restrict__ hlast,
                        unsigned* __restrict__ counter) {
  const int nt = blockIdx.x;
  const int tid = threadIdx.x;
  const int lane = tid & 31, wave = tid >> 5;
  const int g = lane >> 4;

  __shared__ __bf16 sB[32 * 512];        // 32 KB: this n-tile's K=1024 x N=16 block
  __shared__ float  red[4][8][32];       // split-K partials

  {  // stage W_hh block once
    const v16bf* src = (const v16bf*)(pWhh + (size_t)nt * 32 * 512);
    v16bf* dst = (v16bf*)sB;
    for (int c = tid; c < 1024; c += 128) dst[c] = src[c];
  }
  __syncthreads();

  for (int t = 0; t < SS; ++t) {
    v8f acc = {};
    if (t > 0) {
      const float* hbase = Z + (size_t)(t - 1) * HH;   // + m*(S*H) per lane row
#pragma unroll
      for (int q = 0; q < 8; ++q) {
        int kt = wave * 8 + q;
        v16bf a = load_a_f32(hbase, (size_t)SS * HH, kt * 32, lane);
        v16bf b = load_b_pk(sB + kt * 512, lane);      // ds_load_b128 x2
        acc = __builtin_amdgcn_wmma_f32_16x16x32_bf16(false, a, false, b,
                                                      (short)0, acc, false, false);
      }
    }
#pragma unroll
    for (int r = 0; r < 8; ++r) red[wave][r][lane] = acc[r];
    __syncthreads();
    if (wave == 0) {
#pragma unroll
      for (int r = 0; r < 8; ++r) {
        float v = red[0][r][lane] + red[1][r][lane] + red[2][r][lane] + red[3][r][lane];
        int mm = r + 8 * g;                      // batch index 0..15
        int n  = nt * 16 + (lane & 15);
        size_t zi = ((size_t)mm * SS + t) * HH + n;
        float h = tanhf(v + Apre[zi]);
        Z[zi] = h;
        if (t == SS - 1) hlast[mm * HH + n] = h;
      }
    }
    // device-wide barrier (monotonic counter, zeroed by host each launch)
    __threadfence();
    __syncthreads();
    if (tid == 0) {
      atomicAdd(counter, 1u);
      unsigned target = (unsigned)(NWG_R * (t + 1));
      while (__hip_atomic_load(counter, __ATOMIC_ACQUIRE, __HIP_MEMORY_SCOPE_AGENT) < target)
        __builtin_amdgcn_s_sleep(1);
    }
    __syncthreads();
  }
}

// ---------------------------------------------------------------------------
// Phase 3: out = softmax(Z @ W_lin^T + b_lin) per row.
// grid 512 blocks (one 16-row M-tile) x 256 threads (8 waves); wave w -> 4 n-tiles.
// Logits staged in LDS; softmax fully in-block.
// ---------------------------------------------------------------------------
__global__ void k_out(const float* __restrict__ Z, const __bf16* __restrict__ pWlin,
                      const float* __restrict__ b_lin, float* __restrict__ out) {
  const int mt = blockIdx.x;
  const int tid = threadIdx.x;
  const int lane = tid & 31, wave = tid >> 5;
  const int g = lane >> 4;

  __shared__ float sL[16 * DOUT];        // 32 KB logits
  __shared__ float sRed[16 * 16];
  __shared__ float sMax[16], sSum[16];

  v8f acc[4] = {};
  const float* abase = Z + (size_t)(mt * 16) * HH;
  const float* arow  = abase + (size_t)(lane & 15) * HH;
  for (int kt = 0; kt < HH / 32; ++kt) {
    v16bf a = load_a_f32(abase, HH, kt * 32, lane);
    if (kt < HH / 32 - 1)
      __builtin_prefetch(arow + (kt + 1) * 32, 0, 1);
#pragma unroll
    for (int j = 0; j < 4; ++j) {
      int nt = wave * 4 + j;
      v16bf b = load_b_pk(pWlin + (size_t)(nt * (HH / 32) + kt) * 512, lane);
      acc[j] = __builtin_amdgcn_wmma_f32_16x16x32_bf16(false, a, false, b,
                                                       (short)0, acc[j], false, false);
    }
  }
#pragma unroll
  for (int j = 0; j < 4; ++j) {
    int nt = wave * 4 + j;
    int n = nt * 16 + (lane & 15);
    float bias = b_lin[n];
#pragma unroll
    for (int r = 0; r < 8; ++r)
      sL[(r + 8 * g) * DOUT + n] = acc[j][r] + bias;
  }
  __syncthreads();

  const int row16 = tid >> 4, c0 = tid & 15;     // 16 threads per row
  float mx = -3.402823466e38f;
  for (int c = c0; c < DOUT; c += 16) mx = fmaxf(mx, sL[row16 * DOUT + c]);
  sRed[row16 * 16 + c0] = mx;
  __syncthreads();
  if (c0 == 0) {
    float v = sRed[row16 * 16];
#pragma unroll
    for (int i = 1; i < 16; ++i) v = fmaxf(v, sRed[row16 * 16 + i]);
    sMax[row16] = v;
  }
  __syncthreads();
  float sm = 0.f;
  for (int c = c0; c < DOUT; c += 16) {
    float e = __expf(sL[row16 * DOUT + c] - sMax[row16]);
    sL[row16 * DOUT + c] = e;
    sm += e;
  }
  sRed[row16 * 16 + c0] = sm;
  __syncthreads();
  if (c0 == 0) {
    float v = 0.f;
#pragma unroll
    for (int i = 0; i < 16; ++i) v += sRed[row16 * 16 + i];
    sSum[row16] = 1.f / v;
  }
  __syncthreads();
  float inv = sSum[row16];
  size_t rbase = (size_t)(mt * 16 + row16) * DOUT;
  for (int c = c0; c < DOUT; c += 16) out[rbase + c] = sL[row16 * DOUT + c] * inv;
}

// ---------------------------------------------------------------------------
extern "C" void kernel_launch(void* const* d_in, const int* in_sizes, int n_in,
                              void* d_out, int out_size, void* d_ws, size_t ws_size,
                              hipStream_t stream) {
  const float* x     = (const float*)d_in[0];
  // d_in[1] = h0 (zeros by construction; t=0 skips the W_hh matmul)
  const float* W_ih  = (const float*)d_in[2];
  const float* W_hh  = (const float*)d_in[3];
  const float* b_ih  = (const float*)d_in[4];
  const float* b_hh  = (const float*)d_in[5];
  const float* W_lin = (const float*)d_in[6];
  const float* b_lin = (const float*)d_in[7];
  float* out = (float*)d_out;                       // (B,S,DOUT) then (1,B,H)

  char* ws = (char*)d_ws;
  __bf16*   pWih    = (__bf16*)(ws);                          // 2 MB
  __bf16*   pWhh    = (__bf16*)(ws + (2u << 20));             // 2 MB
  __bf16*   pWlin   = (__bf16*)(ws + (4u << 20));             // 1 MB
  float*    Apre    = (float*)(ws + (5u << 20));              // 32 MB
  float*    Z       = (float*)(ws + (5u << 20) + (32u << 20));// 32 MB
  unsigned* counter = (unsigned*)(ws + (5u << 20) + (64u << 20));

  hipMemsetAsync(counter, 0, sizeof(unsigned), stream);

  k_pack_w<<<(HH * DIN / 16 + 255) / 256, 256, 0, stream>>>(W_ih, pWih, HH, DIN);
  k_pack_w<<<(HH * HH  / 16 + 255) / 256, 256, 0, stream>>>(W_hh, pWhh, HH, HH);
  k_pack_w<<<(DOUT * HH / 16 + 255) / 256, 256, 0, stream>>>(W_lin, pWlin, DOUT, HH);

  k_gemm_in<<<ROWS / 16, 256, 0, stream>>>(x, pWih, b_ih, b_hh, Apre);
  k_recur<<<NWG_R, 128, 0, stream>>>(Apre, pWhh, Z, out + (size_t)ROWS * DOUT, counter);
  k_out<<<ROWS / 16, 256, 0, stream>>>(Z, pWlin, b_lin, out);
}
